// BidirGRULayer_66726611911024
// MI455X (gfx1250) — matrix operand order repaired and checked by reading, not compiled, not run on hardware
//
#include <hip/hip_runtime.h>

typedef __attribute__((ext_vector_type(16))) _Float16 v16h;
typedef __attribute__((ext_vector_type(8)))  _Float16 v8h;
typedef __attribute__((ext_vector_type(8)))  float    v8f;

#define T_STEPS 2048
#define BATCH   32
#define IDIM    512
#define HDIM    512
#define KCHUNKS 16   // 512 / 32
#define NTILES  32   // 512 / 16

__device__ __forceinline__ v8f wmma_f16(v16h a, v16h b, v8f c) {
  // D = A(16x32 f16) * B(32x16 f16) + C(16x16 f32)
  return __builtin_amdgcn_wmma_f32_16x16x32_f16(false, a, false, b, (short)0, c,
                                                false, false);
}

__device__ __forceinline__ v16h cat8(v8h lo, v8h hi) {
  return __builtin_shufflevector(lo, hi, 0, 1, 2, 3, 4, 5, 6, 7,
                                 8, 9, 10, 11, 12, 13, 14, 15);
}

// A-fragment (16x32 f16) from a row-major [16][512] f16 LDS tile, chunk kc.
__device__ __forceinline__ v16h lds_afrag(const _Float16* buf, int abase, int kc) {
  v8h lo = *(const v8h*)(buf + abase + kc * 32);
  v8h hi = *(const v8h*)(buf + abase + kc * 32 + 16);
  return cat8(lo, hi);
}

// ---------------------------------------------------------------------------
// Repack fp32 weights (k-major, W[k][n], 512x512) into WMMA B-fragment order:
// pack[(((mim*32 + j)*16 + kc)*32 + lane)*16 + slot]
//   lane 0-15 : n = j*16 + lane,     k = kc*32 + (slot<8 ? slot : 8+slot)      (K 0..7,16..23)
//   lane 16-31: n = j*16 + lane-16,  k = kc*32 + (slot<8 ? 8+slot : 16+slot)   (K 8..15,24..31)
// mim = dir*6 + mi; mi: 0..2 = Wrh,Wch,Wnh (h-part), 3..5 = Wri,Wci,Wni (x-part)
// ---------------------------------------------------------------------------
struct WeightPtrs { const float* w[12]; };

__global__ void repack_weights(WeightPtrs wp, _Float16* pack) {
  int idx = blockIdx.x * 256 + threadIdx.x;
  if (idx >= 12 * 512 * 512) return;
  int slot = idx & 15;
  int lane = (idx >> 4) & 31;
  int kc   = (idx >> 9) & 15;
  int j    = (idx >> 13) & 31;
  int mim  = idx >> 18;
  int n = j * 16 + (lane & 15);
  int k = kc * 32 + (lane < 16 ? 0 : 8) + (slot < 8 ? slot : 8 + slot);
  pack[idx] = (_Float16)wp.w[mim][(size_t)k * HDIM + n];
}

// ---------------------------------------------------------------------------
// Parallel input-projection GEMM (only when ws is large enough):
//   G[(dir*3+g)][s][b][n] = x[t] @ W{r,c,n}i_dir + bias, s = (dir? T-1-t : t)
// grid = (4096, 2), block = 1024 (32 waves). Wave j owns N-tile j, 3 gates.
// ---------------------------------------------------------------------------
struct GateArgs {
  const float*    x;
  const _Float16* pack;
  const float*    bias[6];  // [dir*3+g] = br, bi, bni
  float*          G;
};

__global__ __launch_bounds__(1024) void gate_gemm(GateArgs A) {
  int mt   = blockIdx.x;       // 0..4095 tile of 16 rows of (T*B)
  int dir  = blockIdx.y;
  int tid  = threadIdx.x;
  int lane = tid & 31;
  int j    = tid >> 5;         // wave id == N-tile id

  __shared__ __align__(32) _Float16 x16[16 * IDIM];

  int t  = mt >> 1;
  int b0 = (mt & 1) * 16;
  const float* xrow = A.x + ((size_t)t * BATCH + b0) * IDIM;
  for (int e = 0; e < 8; ++e) {
    int i = tid * 8 + e;                 // 16 rows are contiguous in memory
    x16[i] = (_Float16)xrow[i];
  }

  int nl    = lane & 15;
  int m_add = (lane < 16) ? 0 : 8;
  int abase = ((lane < 16) ? lane : lane - 16) * IDIM + ((lane < 16) ? 0 : 8);
  int s     = (dir == 0) ? t : (T_STEPS - 1 - t);
  const size_t gstep = (size_t)T_STEPS * BATCH * HDIM;

  // hoist bias + weight pointers before the barrier (kernarg loads once)
  float bv0 = A.bias[dir * 3 + 0][j * 16 + nl];
  float bv1 = A.bias[dir * 3 + 1][j * 16 + nl];
  float bv2 = A.bias[dir * 3 + 2][j * 16 + nl];
  const _Float16* bp0 =
      A.pack + ((((size_t)dir * 6 + 3) * NTILES + j) * KCHUNKS) * 512 + lane * 16;
  const _Float16* bp1 = bp0 + (size_t)NTILES * KCHUNKS * 512;
  const _Float16* bp2 = bp1 + (size_t)NTILES * KCHUNKS * 512;

  v8f a0, a1, a2;
  for (int v = 0; v < 8; ++v) { a0[v] = bv0; a1[v] = bv1; a2[v] = bv2; }

  __syncthreads();

  v16h w0 = *(const v16h*)bp0;
  v16h w1 = *(const v16h*)bp1;
  v16h w2 = *(const v16h*)bp2;
#pragma unroll
  for (int kc = 0; kc < KCHUNKS; ++kc) {
    v16h n0, n1, n2;
    if (kc < KCHUNKS - 1) {
      n0 = *(const v16h*)(bp0 + (size_t)(kc + 1) * 512);
      n1 = *(const v16h*)(bp1 + (size_t)(kc + 1) * 512);
      n2 = *(const v16h*)(bp2 + (size_t)(kc + 1) * 512);
    }
    v16h a = lds_afrag(x16, abase, kc);
    a0 = wmma_f16(a, w0, a0);
    a1 = wmma_f16(a, w1, a1);
    a2 = wmma_f16(a, w2, a2);
    w0 = n0; w1 = n1; w2 = n2;
  }

  float* gbase = A.G + (size_t)(dir * 3) * gstep + (size_t)s * BATCH * HDIM;
  for (int v = 0; v < 8; ++v) {
    size_t off = (size_t)(b0 + v + m_add) * HDIM + j * 16 + nl;
    gbase[off]             = a0[v];
    gbase[gstep + off]     = a1[v];
    gbase[2 * gstep + off] = a2[v];
  }
}

// ---------------------------------------------------------------------------
// Sequential bidirectional GRU scan. grid = (2 dirs, 2 batch-halves), 1024 thr.
// wave j owns output N-tile j: keeps accR/accC/accNX/accNH in registers,
// combines with sigmoid/tanh locally, no LDS gate exchange needed.
// ---------------------------------------------------------------------------
struct RecArgs {
  const float*    x;
  const _Float16* pack;
  const float*    G;        // nullptr => fused x-projection
  const float*    br[2];
  const float*    bi[2];
  const float*    bni[2];
  const float*    bnh[2];
  float*          out;
  int             fused;
};

__global__ __launch_bounds__(1024) void bigru_scan(RecArgs A) {
  int dir  = blockIdx.x;
  int b0   = blockIdx.y * 16;
  int tid  = threadIdx.x;
  int lane = tid & 31;
  int j    = tid >> 5;          // wave id == N-tile id (0..31)

  __shared__ __align__(32) _Float16 h16[16 * HDIM];
  __shared__ __align__(32) _Float16 x16[16 * IDIM];

  for (int e = 0; e < 8; ++e) h16[tid * 8 + e] = (_Float16)0.f;

  int nl     = lane & 15;
  int nglob  = j * 16 + nl;
  float b_r  = A.br[dir][nglob];
  float b_i  = A.bi[dir][nglob];
  float b_ni = A.bni[dir][nglob];
  float b_nh = A.bnh[dir][nglob];

  // weight base pointers, h-part (0..2) and x-part (3..5)
  const _Float16* wh0 =
      A.pack + ((((size_t)dir * 6 + 0) * NTILES + j) * KCHUNKS) * 512 + lane * 16;
  const size_t mstep = (size_t)NTILES * KCHUNKS * 512;
  const _Float16* wh1 = wh0 + mstep;
  const _Float16* wh2 = wh1 + mstep;
  const _Float16* wx0 = wh2 + mstep;
  const _Float16* wx1 = wx0 + mstep;
  const _Float16* wx2 = wx1 + mstep;

  int m_add = (lane < 16) ? 0 : 8;
  int abase = ((lane < 16) ? lane : lane - 16) * HDIM + ((lane < 16) ? 0 : 8);
  const size_t gstep = (size_t)T_STEPS * BATCH * HDIM;
  const int fused = A.fused;

  for (int s = 0; s < T_STEPS; ++s) {
    int tt = (dir == 0) ? s : (T_STEPS - 1 - s);

    if (fused) {  // stage x_t tile as f16 A-source
      const float* xrow = A.x + ((size_t)tt * BATCH + b0) * IDIM;
      for (int e = 0; e < 8; ++e) {
        int i = tid * 8 + e;
        x16[i] = (_Float16)xrow[i];
      }
    }
    __syncthreads();  // h16 (prev step) and x16 ready

    v8f accR, accC, accNX, accNH;
    if (fused) {
      for (int v = 0; v < 8; ++v) {
        accR[v] = b_r; accC[v] = b_i; accNX[v] = b_ni; accNH[v] = b_nh;
      }
    } else {
      const float* g0 = A.G + (size_t)(dir * 3) * gstep + (size_t)s * BATCH * HDIM;
      for (int v = 0; v < 8; ++v) {
        size_t off = (size_t)(b0 + v + m_add) * HDIM + nglob;
        accR[v]  = g0[off];
        accC[v]  = g0[gstep + off];
        accNX[v] = g0[2 * gstep + off];
        accNH[v] = b_nh;
      }
    }

    // ---- pass 1: h-part (double-buffered B stream) ----
    {
      v16h w0 = *(const v16h*)wh0;
      v16h w1 = *(const v16h*)wh1;
      v16h w2 = *(const v16h*)wh2;
#pragma unroll
      for (int kc = 0; kc < KCHUNKS; ++kc) {
        v16h n0, n1, n2;
        if (kc < KCHUNKS - 1) {
          n0 = *(const v16h*)(wh0 + (size_t)(kc + 1) * 512);
          n1 = *(const v16h*)(wh1 + (size_t)(kc + 1) * 512);
          n2 = *(const v16h*)(wh2 + (size_t)(kc + 1) * 512);
        }
        v16h ah = lds_afrag(h16, abase, kc);
        accR  = wmma_f16(ah, w0, accR);
        accC  = wmma_f16(ah, w1, accC);
        accNH = wmma_f16(ah, w2, accNH);
        w0 = n0; w1 = n1; w2 = n2;
      }
    }
    // ---- pass 2: x-part (fused path only) ----
    if (fused) {
      v16h w0 = *(const v16h*)wx0;
      v16h w1 = *(const v16h*)wx1;
      v16h w2 = *(const v16h*)wx2;
#pragma unroll
      for (int kc = 0; kc < KCHUNKS; ++kc) {
        v16h n0, n1, n2;
        if (kc < KCHUNKS - 1) {
          n0 = *(const v16h*)(wx0 + (size_t)(kc + 1) * 512);
          n1 = *(const v16h*)(wx1 + (size_t)(kc + 1) * 512);
          n2 = *(const v16h*)(wx2 + (size_t)(kc + 1) * 512);
        }
        v16h ax = lds_afrag(x16, abase, kc);
        accR  = wmma_f16(ax, w0, accR);
        accC  = wmma_f16(ax, w1, accC);
        accNX = wmma_f16(ax, w2, accNX);
        w0 = n0; w1 = n1; w2 = n2;
      }
    }

    // elementwise GRU combine, all in registers (C-layout: row v+m_add, col nl)
    float hnew[8];
    for (int v = 0; v < 8; ++v) {
      float r = 1.f / (1.f + expf(-accR[v]));
      float c = 1.f / (1.f + expf(-accC[v]));
      float n = tanhf(accNX[v] + r * accNH[v]);
      float hp = (float)h16[(v + m_add) * HDIM + nglob];
      hnew[v] = n + c * (hp - n);
    }
    __syncthreads();  // everyone done reading h16

    float* orow = A.out + (size_t)tt * BATCH * (2 * HDIM) + (size_t)dir * HDIM;
    for (int v = 0; v < 8; ++v) {
      int m = v + m_add;
      h16[(size_t)m * HDIM + nglob] = (_Float16)hnew[v];
      orow[(size_t)(b0 + m) * (2 * HDIM) + nglob] = hnew[v];
    }
  }
}

// ---------------------------------------------------------------------------
extern "C" void kernel_launch(void* const* d_in, const int* in_sizes, int n_in,
                              void* d_out, int out_size, void* d_ws, size_t ws_size,
                              hipStream_t stream) {
  (void)in_sizes; (void)n_in; (void)out_size;

  // d_in: 0=x; per dir base=1+dir*10: +0 Wri,+1 Wci,+2 Wni,+3 Wrh,+4 Wch,+5 Wnh,
  //                                    +6 br,+7 bi,+8 bni,+9 bnh
  WeightPtrs wpk;
  for (int dir = 0; dir < 2; ++dir) {
    int base = 1 + dir * 10;
    wpk.w[dir * 6 + 0] = (const float*)d_in[base + 3];  // Wrh
    wpk.w[dir * 6 + 1] = (const float*)d_in[base + 4];  // Wch
    wpk.w[dir * 6 + 2] = (const float*)d_in[base + 5];  // Wnh
    wpk.w[dir * 6 + 3] = (const float*)d_in[base + 0];  // Wri
    wpk.w[dir * 6 + 4] = (const float*)d_in[base + 1];  // Wci
    wpk.w[dir * 6 + 5] = (const float*)d_in[base + 2];  // Wni
  }

  _Float16* pack = (_Float16*)d_ws;
  size_t packBytes = (size_t)12 * 512 * 512 * sizeof(_Float16);          // 6 MiB
  size_t packPad   = (packBytes + 255) & ~(size_t)255;
  size_t gBytes    = (size_t)6 * T_STEPS * BATCH * HDIM * sizeof(float); // 768 MiB
  bool big = ws_size >= packPad + gBytes;
  float* G = (float*)((char*)d_ws + packPad);

  repack_weights<<<dim3((12 * 512 * 512 + 255) / 256), dim3(256), 0, stream>>>(
      wpk, pack);

  if (big) {
    GateArgs ga;
    ga.x = (const float*)d_in[0];
    ga.pack = pack;
    ga.G = G;
    for (int dir = 0; dir < 2; ++dir) {
      int base = 1 + dir * 10;
      ga.bias[dir * 3 + 0] = (const float*)d_in[base + 6];  // br
      ga.bias[dir * 3 + 1] = (const float*)d_in[base + 7];  // bi
      ga.bias[dir * 3 + 2] = (const float*)d_in[base + 8];  // bni
    }
    gate_gemm<<<dim3(T_STEPS * BATCH / 16, 2), dim3(1024), 0, stream>>>(ga);
  }

  RecArgs ra;
  ra.x = (const float*)d_in[0];
  ra.pack = pack;
  ra.G = big ? G : nullptr;
  ra.fused = big ? 0 : 1;
  for (int dir = 0; dir < 2; ++dir) {
    int base = 1 + dir * 10;
    ra.br[dir]  = (const float*)d_in[base + 6];
    ra.bi[dir]  = (const float*)d_in[base + 7];
    ra.bni[dir] = (const float*)d_in[base + 8];
    ra.bnh[dir] = (const float*)d_in[base + 9];
  }
  ra.out = (float*)d_out;

  bigru_scan<<<dim3(2, 2), dim3(1024), 0, stream>>>(ra);
}
